// EncoderBlock_10445360464289
// MI455X (gfx1250) — compile-verified, matrix operand order checked by reading, not run on hardware
//
#include <hip/hip_runtime.h>
#include <hip/hip_bf16.h>

// ---------------------------------------------------------------------------
// Transformer encoder block for MI455X (gfx1250, wave32, WMMA).
// bf16 inputs / f32 accumulate via v_wmma_f32_16x16x32_bf16.
// GEMM staging uses CDNA5 async global->LDS (ASYNCcnt) with double buffering.
// ---------------------------------------------------------------------------

typedef __bf16 bf16_t;
typedef __attribute__((ext_vector_type(16))) __bf16 bf16x16;
typedef __attribute__((ext_vector_type(8)))  __bf16 bf16x8;
typedef __attribute__((ext_vector_type(8)))  float  f32x8;

#define D_MODEL 1024
#define NHEAD   16
#define DKH     64
#define D_FF    4096
#define SEQ     2048
#define BATCH   2
#define ROWS    (BATCH * SEQ)   // 4096

__device__ __forceinline__ f32x8 zero8() {
  f32x8 z;
#pragma unroll
  for (int i = 0; i < 8; ++i) z[i] = 0.0f;
  return z;
}

// 16-byte async copy global -> LDS (CDNA5 GLOBAL_LOAD_ASYNC_TO_LDS_B128,
// tracked by ASYNCcnt). lds_off is the 32-bit LDS byte offset (generic
// pointers to __shared__ truncate to the LDS offset per the aperture rules).
__device__ __forceinline__ void async_copy16(unsigned lds_off, const bf16_t* gptr) {
  asm volatile("global_load_async_to_lds_b128 %0, %1, off"
               :: "v"(lds_off), "v"(gptr)
               : "memory");
}
__device__ __forceinline__ void wait_async0() {
  asm volatile("s_wait_asynccnt 0x0" ::: "memory");
}
__device__ __forceinline__ unsigned lds_off_of(const bf16_t* p) {
  return (unsigned)(size_t)p;
}

// Build a 16-element bf16 fragment from two contiguous 8-element (16-byte) LDS loads.
// ISA 16-bit A/B layout: lane group g=0 -> K {0..7,16..23}; g=1 -> K {8..15,24..31}.
__device__ __forceinline__ bf16x16 ld_frag(const bf16_t* p_lo, const bf16_t* p_hi) {
  bf16x8 lo = *(const bf16x8*)p_lo;
  bf16x8 hi = *(const bf16x8*)p_hi;
  bf16x16 r;
#pragma unroll
  for (int e = 0; e < 8; ++e) { r[e] = lo[e]; r[8 + e] = hi[e]; }
  return r;
}

// ---------------------------------------------------------------------------
// f32 [K][N]  ->  bf16 [N][K]   (weights pre-transposed so GEMM B-tiles are
// loaded exactly like A-tiles: contiguous along the contraction dim)
// ---------------------------------------------------------------------------
__global__ __launch_bounds__(256) void k_transpose_bf16(
    const float* __restrict__ in, bf16_t* __restrict__ out, int Kd, int Nd) {
  int idx = blockIdx.x * 256 + threadIdx.x;
  if (idx >= Kd * Nd) return;
  int k = idx / Nd, n = idx - k * Nd;
  out[(size_t)n * Kd + k] = (bf16_t)in[idx];
}

// ---------------------------------------------------------------------------
// LayerNorm, faithful to reference: unbiased std (ddof=1), eps added to std.
// f32 in -> bf16 out. One block per row of 1024.
// ---------------------------------------------------------------------------
__global__ __launch_bounds__(256) void k_layernorm_bf16(
    const float* __restrict__ x, const float* __restrict__ alpha,
    const float* __restrict__ beta, bf16_t* __restrict__ out) {
  const int row = blockIdx.x;
  const float* xr = x + (size_t)row * D_MODEL;
  float s = 0.f, ss = 0.f;
  for (int i = threadIdx.x; i < D_MODEL; i += 256) {
    float v = xr[i]; s += v; ss += v * v;
  }
  __shared__ float r1[256], r2[256];
  r1[threadIdx.x] = s; r2[threadIdx.x] = ss;
  __syncthreads();
  for (int o = 128; o > 0; o >>= 1) {
    if (threadIdx.x < o) { r1[threadIdx.x] += r1[threadIdx.x + o];
                           r2[threadIdx.x] += r2[threadIdx.x + o]; }
    __syncthreads();
  }
  float mean = r1[0] * (1.0f / D_MODEL);
  float var  = (r2[0] - (float)D_MODEL * mean * mean) * (1.0f / (D_MODEL - 1));
  var = var > 0.f ? var : 0.f;
  float scale = alpha[0] / (sqrtf(var) + 1e-5f);
  float shift = beta[0];
  bf16_t* orow = out + (size_t)row * D_MODEL;
  for (int i = threadIdx.x; i < D_MODEL; i += 256)
    orow[i] = (bf16_t)((xr[i] - mean) * scale + shift);
}

// ---------------------------------------------------------------------------
// WMMA GEMM: C[M,N] = A[M,K] @ B[K,N], with B supplied transposed (Bt[N][K]).
// 128x128 block tile, BK=32, 256 threads = 8 waves arranged 2(M) x 4(N);
// each wave computes 64x32 = 4x2 accumulator tiles -> 8 WMMA per k-step.
// Double-buffered async global->LDS staging: tile kt+1 streams while tile kt
// is consumed. One barrier per k-step.
// Epilogue fuses bias / ReLU / residual; writes f32 and/or bf16.
// ---------------------------------------------------------------------------
__global__ __launch_bounds__(256) void k_gemm_bf16_wmma(
    const bf16_t* __restrict__ A, const bf16_t* __restrict__ Bt,
    const float* __restrict__ bias, const float* __restrict__ residual,
    float* __restrict__ Cf, bf16_t* __restrict__ Cb,
    int M, int N, int K, int doRelu) {
  __shared__ __align__(16) bf16_t As[2][128 * 32];
  __shared__ __align__(16) bf16_t Bs[2][128 * 32];
  const int tid  = threadIdx.x;
  const int wave = tid >> 5, lane = tid & 31, lh = lane & 15, g = lane >> 4;
  const int waveM = wave >> 2, waveN = wave & 3;       // 2 x 4 wave grid
  const int rowBase = blockIdx.y * 128;
  const int colBase = blockIdx.x * 128;
  const int kOffLo = g ? 8 : 0;
  const int kOffHi = g ? 24 : 16;

  // Per-thread staging slots: rows r0 and r0+64, 8 bf16 starting at col c.
  const int r0 = tid >> 2;
  const int c0 = (tid & 3) * 8;

  auto stage = [&](int kt, int buf) {
    const int kcol = kt * 32 + c0;
    async_copy16(lds_off_of(&As[buf][r0 * 32 + c0]),
                 &A[(size_t)(rowBase + r0) * K + kcol]);
    async_copy16(lds_off_of(&As[buf][(r0 + 64) * 32 + c0]),
                 &A[(size_t)(rowBase + r0 + 64) * K + kcol]);
    async_copy16(lds_off_of(&Bs[buf][r0 * 32 + c0]),
                 &Bt[(size_t)(colBase + r0) * K + kcol]);
    async_copy16(lds_off_of(&Bs[buf][(r0 + 64) * 32 + c0]),
                 &Bt[(size_t)(colBase + r0 + 64) * K + kcol]);
  };

  f32x8 acc[4][2];
#pragma unroll
  for (int mr = 0; mr < 4; ++mr)
#pragma unroll
    for (int nr = 0; nr < 2; ++nr) acc[mr][nr] = zero8();

  const int kTiles = K >> 5;
  stage(0, 0);                               // prologue prefetch
  for (int kt = 0; kt < kTiles; ++kt) {
    const int cur = kt & 1;
    wait_async0();                           // own async writes to buf(cur) done
    __syncthreads();                         // all waves: buf(cur) ready, prev reads retired
    if (kt + 1 < kTiles) stage(kt + 1, cur ^ 1);   // overlap with compute below

    bf16x16 afrag[4], bfrag[2];
#pragma unroll
    for (int mr = 0; mr < 4; ++mr) {
      int r = waveM * 64 + mr * 16 + lh;
      afrag[mr] = ld_frag(&As[cur][r * 32 + kOffLo], &As[cur][r * 32 + kOffHi]);
    }
#pragma unroll
    for (int nr = 0; nr < 2; ++nr) {
      int r = waveN * 32 + nr * 16 + lh;
      bfrag[nr] = ld_frag(&Bs[cur][r * 32 + kOffLo], &Bs[cur][r * 32 + kOffHi]);
    }
#pragma unroll
    for (int mr = 0; mr < 4; ++mr)
#pragma unroll
      for (int nr = 0; nr < 2; ++nr)
        acc[mr][nr] = __builtin_amdgcn_wmma_f32_16x16x32_bf16(
            false, afrag[mr], false, bfrag[nr], (short)0, acc[mr][nr],
            false, false);
  }

  // Epilogue. C-frag layout: lane L holds rows 8*(L/16)+r, column L%16.
#pragma unroll
  for (int mr = 0; mr < 4; ++mr) {
#pragma unroll
    for (int nr = 0; nr < 2; ++nr) {
      int col = colBase + waveN * 32 + nr * 16 + lh;
#pragma unroll
      for (int r = 0; r < 8; ++r) {
        int row = rowBase + waveM * 64 + mr * 16 + g * 8 + r;
        float v = acc[mr][nr][r];
        if (bias)     v += bias[col];
        if (doRelu)   v  = v > 0.f ? v : 0.f;
        if (residual) v += residual[(size_t)row * N + col];
        if (Cf) Cf[(size_t)row * N + col] = v;
        if (Cb) Cb[(size_t)row * N + col] = (bf16_t)v;
      }
    }
  }
}

// ---------------------------------------------------------------------------
// Flash attention: one block per (64-query tile, head, batch). 4 waves; each
// wave owns 16 query rows. Streams 64-key K/V tiles through LDS with online
// softmax; P restaged via LDS for the PV matmul (A-frag layout); V transposed
// at load so V columns are contiguous B-fragments. Q/K tiles use the async
// global->LDS path; V needs the in-flight transpose so it stays on vector loads.
// ---------------------------------------------------------------------------
__global__ __launch_bounds__(128) void k_attention(
    const bf16_t* __restrict__ Q, const bf16_t* __restrict__ Kmat,
    const bf16_t* __restrict__ V, const int* __restrict__ mask,
    bf16_t* __restrict__ ctx) {
  __shared__ __align__(16) bf16_t Qs[64 * 64];
  __shared__ __align__(16) bf16_t Ks[64 * 64];
  __shared__ __align__(16) bf16_t Vt[64 * 64];   // Vt[d][key]
  __shared__ __align__(16) bf16_t Ps[64 * 64];
  const int tid  = threadIdx.x;
  const int wave = tid >> 5, lane = tid & 31, lh = lane & 15, g = lane >> 4;
  const int qbase = blockIdx.x * 64;
  const int h = blockIdx.y, b = blockIdx.z;
  const size_t row0 = (size_t)b * SEQ;
  const int col0 = h * DKH;
  const int kOffLo = g ? 8 : 0;
  const int kOffHi = g ? 24 : 16;

  // Load Q tile [64 x 64] via async copies.
#pragma unroll
  for (int i = 0; i < 4; ++i) {
    int id = i * 128 + tid;                // 0..511
    int r = id >> 3, c = (id & 7) * 8;
    async_copy16(lds_off_of(&Qs[r * 64 + c]),
                 &Q[(row0 + qbase + r) * D_MODEL + col0 + c]);
  }
  wait_async0();
  __syncthreads();

  // Q A-fragments are invariant across key tiles: hoist.
  bf16x16 aq[2];
#pragma unroll
  for (int c = 0; c < 2; ++c) {
    int r = wave * 16 + lh;
    aq[c] = ld_frag(&Qs[r * 64 + c * 32 + kOffLo],
                    &Qs[r * 64 + c * 32 + kOffHi]);
  }

  f32x8 o[4];
#pragma unroll
  for (int nr = 0; nr < 4; ++nr) o[nr] = zero8();
  float m[8], l[8];
#pragma unroll
  for (int r = 0; r < 8; ++r) { m[r] = -1e30f; l[r] = 0.f; }
  const float sm_scale = 0.125f;           // 1/sqrt(64)

  for (int kt = 0; kt < SEQ / 64; ++kt) {
    const int kbase = kt * 64;
    __syncthreads();                       // prior reads of Ks/Vt/Ps retired
    // K tile: straight async copy (rows=key already matches B-frag layout).
    // V tile: vector load + transpose scatter into Vt[d][key].
#pragma unroll
    for (int i = 0; i < 4; ++i) {
      int id = i * 128 + tid;
      int r = id >> 3, c = (id & 7) * 8;
      async_copy16(lds_off_of(&Ks[r * 64 + c]),
                   &Kmat[(row0 + kbase + r) * D_MODEL + col0 + c]);
      uint4 vv4 =
          *(const uint4*)(&V[(row0 + kbase + r) * D_MODEL + col0 + c]);
      const bf16_t* vv = (const bf16_t*)&vv4;
#pragma unroll
      for (int j = 0; j < 8; ++j) Vt[(c + j) * 64 + r] = vv[j];
    }
    wait_async0();
    __syncthreads();                       // Ks async + Vt ds stores visible

    // Scores: wave's 16 q-rows x 64 keys, contraction over d=64 (2 chunks).
    f32x8 s[4];
#pragma unroll
    for (int nr = 0; nr < 4; ++nr) {
      s[nr] = zero8();
#pragma unroll
      for (int c = 0; c < 2; ++c) {
        int kr = nr * 16 + lh;
        bf16x16 bk = ld_frag(&Ks[kr * 64 + c * 32 + kOffLo],
                             &Ks[kr * 64 + c * 32 + kOffHi]);
        s[nr] = __builtin_amdgcn_wmma_f32_16x16x32_bf16(
            false, aq[c], false, bk, (short)0, s[nr], false, false);
      }
    }
    // Scale + mask (mask shape [B,1,1,S], key-indexed).
#pragma unroll
    for (int nr = 0; nr < 4; ++nr) {
      int key = kbase + nr * 16 + lh;
      bool live = mask[row0 + key] != 0;
#pragma unroll
      for (int r = 0; r < 8; ++r) {
        float sv = s[nr][r] * sm_scale;
        s[nr][r] = live ? sv : -1e9f;
      }
    }
    // Online softmax: row reductions across the 16-lane half-wave.
    float f[8];
#pragma unroll
    for (int r = 0; r < 8; ++r) {
      float v = fmaxf(fmaxf(s[0][r], s[1][r]), fmaxf(s[2][r], s[3][r]));
#pragma unroll
      for (int off = 8; off > 0; off >>= 1) v = fmaxf(v, __shfl_xor(v, off, 16));
      float mn = fmaxf(m[r], v);
      f[r] = __expf(m[r] - mn);
      m[r] = mn;
    }
    float rs[8];
#pragma unroll
    for (int r = 0; r < 8; ++r) rs[r] = 0.f;
#pragma unroll
    for (int nr = 0; nr < 4; ++nr)
#pragma unroll
      for (int r = 0; r < 8; ++r) {
        float p = __expf(s[nr][r] - m[r]);
        s[nr][r] = p; rs[r] += p;
      }
#pragma unroll
    for (int r = 0; r < 8; ++r) {
      float v = rs[r];
#pragma unroll
      for (int off = 8; off > 0; off >>= 1) v += __shfl_xor(v, off, 16);
      l[r] = l[r] * f[r] + v;
    }
    // Rescale running context, stage P into LDS (C-frag -> A-frag transpose).
#pragma unroll
    for (int nr = 0; nr < 4; ++nr)
#pragma unroll
      for (int r = 0; r < 8; ++r) o[nr][r] *= f[r];
#pragma unroll
    for (int nr = 0; nr < 4; ++nr)
#pragma unroll
      for (int r = 0; r < 8; ++r)
        Ps[(wave * 16 + g * 8 + r) * 64 + nr * 16 + lh] = (bf16_t)s[nr][r];
    __syncthreads();

    // ctx += P @ V  (contraction over 64 keys, 2 chunks).
#pragma unroll
    for (int c = 0; c < 2; ++c) {
      int pr = wave * 16 + lh;
      bf16x16 ap = ld_frag(&Ps[pr * 64 + c * 32 + kOffLo],
                           &Ps[pr * 64 + c * 32 + kOffHi]);
#pragma unroll
      for (int nr = 0; nr < 4; ++nr) {
        int dr = nr * 16 + lh;
        bf16x16 bv = ld_frag(&Vt[dr * 64 + c * 32 + kOffLo],
                             &Vt[dr * 64 + c * 32 + kOffHi]);
        o[nr] = __builtin_amdgcn_wmma_f32_16x16x32_bf16(
            false, ap, false, bv, (short)0, o[nr], false, false);
      }
    }
  }

  // Normalize and scatter ctx back to [B,S,H*Dk] layout (bf16).
#pragma unroll
  for (int nr = 0; nr < 4; ++nr)
#pragma unroll
    for (int r = 0; r < 8; ++r) {
      float v = o[nr][r] / l[r];
      ctx[(row0 + qbase + wave * 16 + g * 8 + r) * D_MODEL + col0 + nr * 16 + lh]
          = (bf16_t)v;
    }
}

// ---------------------------------------------------------------------------
// Host-side orchestration (graph-capture safe: stream launches only).
// ---------------------------------------------------------------------------
extern "C" void kernel_launch(void* const* d_in, const int* in_sizes, int n_in,
                              void* d_out, int out_size, void* d_ws, size_t ws_size,
                              hipStream_t stream) {
  const float* x      = (const float*)d_in[0];
  const int*   mask   = (const int*)d_in[1];
  const float* wq     = (const float*)d_in[2];
  const float* wk     = (const float*)d_in[3];
  const float* wv     = (const float*)d_in[4];
  const float* wo     = (const float*)d_in[5];
  const float* w1     = (const float*)d_in[6];
  const float* b1     = (const float*)d_in[7];
  const float* w2     = (const float*)d_in[8];
  const float* b2     = (const float*)d_in[9];
  const float* alpha1 = (const float*)d_in[10];
  const float* bias1  = (const float*)d_in[11];
  const float* alpha2 = (const float*)d_in[12];
  const float* bias2  = (const float*)d_in[13];
  float* out = (float*)d_out;

  // Workspace carve-up (~120 MB), 256B-aligned slices.
  char* ws = (char*)d_ws;
  size_t off = 0;
  auto alloc = [&](size_t bytes) -> char* {
    char* p = ws + off;
    off += (bytes + 255) & ~(size_t)255;
    return p;
  };
  bf16_t* wqT = (bf16_t*)alloc((size_t)D_MODEL * D_MODEL * 2);
  bf16_t* wkT = (bf16_t*)alloc((size_t)D_MODEL * D_MODEL * 2);
  bf16_t* wvT = (bf16_t*)alloc((size_t)D_MODEL * D_MODEL * 2);
  bf16_t* woT = (bf16_t*)alloc((size_t)D_MODEL * D_MODEL * 2);
  bf16_t* w1T = (bf16_t*)alloc((size_t)D_FF * D_MODEL * 2);   // [4096][1024]
  bf16_t* w2T = (bf16_t*)alloc((size_t)D_MODEL * D_FF * 2);   // [1024][4096]
  bf16_t* xn  = (bf16_t*)alloc((size_t)ROWS * D_MODEL * 2);
  bf16_t* qb  = (bf16_t*)alloc((size_t)ROWS * D_MODEL * 2);
  bf16_t* kb  = (bf16_t*)alloc((size_t)ROWS * D_MODEL * 2);
  bf16_t* vb  = (bf16_t*)alloc((size_t)ROWS * D_MODEL * 2);
  bf16_t* ctxb= (bf16_t*)alloc((size_t)ROWS * D_MODEL * 2);
  float*  x2  = (float*) alloc((size_t)ROWS * D_MODEL * 4);
  bf16_t* xn2 = (bf16_t*)alloc((size_t)ROWS * D_MODEL * 2);
  bf16_t* hb  = (bf16_t*)alloc((size_t)ROWS * D_FF * 2);
  (void)ws_size; (void)in_sizes; (void)n_in; (void)out_size;

  // 1) Weights -> bf16, transposed to [N][K].
  k_transpose_bf16<<<(D_MODEL * D_MODEL + 255) / 256, 256, 0, stream>>>(wq, wqT, D_MODEL, D_MODEL);
  k_transpose_bf16<<<(D_MODEL * D_MODEL + 255) / 256, 256, 0, stream>>>(wk, wkT, D_MODEL, D_MODEL);
  k_transpose_bf16<<<(D_MODEL * D_MODEL + 255) / 256, 256, 0, stream>>>(wv, wvT, D_MODEL, D_MODEL);
  k_transpose_bf16<<<(D_MODEL * D_MODEL + 255) / 256, 256, 0, stream>>>(wo, woT, D_MODEL, D_MODEL);
  k_transpose_bf16<<<(D_MODEL * D_FF + 255) / 256, 256, 0, stream>>>(w1, w1T, D_MODEL, D_FF);
  k_transpose_bf16<<<(D_FF * D_MODEL + 255) / 256, 256, 0, stream>>>(w2, w2T, D_FF, D_MODEL);

  // 2) LN1
  k_layernorm_bf16<<<ROWS, 256, 0, stream>>>(x, alpha1, bias1, xn);

  // 3) Q/K/V projections (bf16 out)
  dim3 gProj(D_MODEL / 128, ROWS / 128);
  k_gemm_bf16_wmma<<<gProj, 256, 0, stream>>>(xn, wqT, nullptr, nullptr, nullptr, qb, ROWS, D_MODEL, D_MODEL, 0);
  k_gemm_bf16_wmma<<<gProj, 256, 0, stream>>>(xn, wkT, nullptr, nullptr, nullptr, kb, ROWS, D_MODEL, D_MODEL, 0);
  k_gemm_bf16_wmma<<<gProj, 256, 0, stream>>>(xn, wvT, nullptr, nullptr, nullptr, vb, ROWS, D_MODEL, D_MODEL, 0);

  // 4) Flash attention
  dim3 gAttn(SEQ / 64, NHEAD, BATCH);
  k_attention<<<gAttn, 128, 0, stream>>>(qb, kb, vb, mask, ctxb);

  // 5) Output projection + residual (f32 out)
  k_gemm_bf16_wmma<<<gProj, 256, 0, stream>>>(ctxb, woT, nullptr, x, x2, nullptr, ROWS, D_MODEL, D_MODEL, 0);

  // 6) LN2
  k_layernorm_bf16<<<ROWS, 256, 0, stream>>>(x2, alpha2, bias2, xn2);

  // 7) FFN up: relu(xn2 @ w1 + b1) -> bf16
  dim3 gFF1(D_FF / 128, ROWS / 128);
  k_gemm_bf16_wmma<<<gFF1, 256, 0, stream>>>(xn2, w1T, b1, nullptr, nullptr, hb, ROWS, D_FF, D_MODEL, 1);

  // 8) FFN down + bias + residual -> d_out (f32)
  dim3 gFF2(D_MODEL / 128, ROWS / 128);
  k_gemm_bf16_wmma<<<gFF2, 256, 0, stream>>>(hb, w2T, b2, x2, out, ROWS ? nullptr : nullptr, ROWS, D_MODEL, D_FF, 0);
}
